// CostVolume_57406532878563
// MI455X (gfx1250) — matrix operand order repaired and checked by reading, compiled, not run on hardware
//
#include <hip/hip_runtime.h>

typedef __attribute__((ext_vector_type(16))) _Float16 v16h;
typedef __attribute__((ext_vector_type(8)))  _Float16 v8h;
typedef __attribute__((ext_vector_type(8)))  float    v8f;

#define B_    4
#define C_    128
#define H_    128
#define W_    256
#define WP_   (W_ + 32)   // f2 padded width: 16 zero cols each side
#define ND    9           // 2*D+1
#define NDISP 81          // ND*ND
#define BANDW (ND * W_ + 32)   // one dy-band + per-lane dump slots

// ---------------------------------------------------------------------------
// Zero the 16-column left/right pads of one padded f2 row per block.
// ---------------------------------------------------------------------------
__global__ __launch_bounds__(256) void cv_pad_kernel(_Float16* __restrict__ ws2) {
  const int row = blockIdx.x;               // b*H + h
  const int t   = threadIdx.x;
  const size_t rowbase = (size_t)row * (WP_ * C_);
  v16h z = {};
  size_t off;
  if (t < 128) off = rowbase + (size_t)t * 16;                                  // cols [0,16)
  else         off = rowbase + (size_t)(W_ + 16) * C_ + (size_t)(t - 128) * 16; // cols [272,288)
  *(v16h*)(ws2 + off) = z;
}

// ---------------------------------------------------------------------------
// Stage 1: L2-normalize over channels. NCHW f32 -> channel-contiguous f16 with
// parameterized row pitch (wpc) and column offset (coff): writes f1 (W,0) and
// the padded f2 (288,16) with the same kernel.
// ---------------------------------------------------------------------------
__global__ __launch_bounds__(256) void cv_norm_kernel(const float* __restrict__ in,
                                                      _Float16* __restrict__ outh,
                                                      int wpc, int coff) {
  const int HW = H_ * W_;
  int p = blockIdx.x * 256 + threadIdx.x;   // pixel over B*H*W (exact grid)
  int b = p / HW;
  int s = p - b * HW;
  int h = s >> 8;                           // W_ == 256
  int w = s & (W_ - 1);
  const float* base = in + (size_t)b * C_ * HW + s;

  float ss = 0.f;
#pragma unroll 8
  for (int c = 0; c < C_; ++c) {
    float v = base[(size_t)c * HW];
    ss += v * v;
  }
  float inv = 1.0f / fmaxf(sqrtf(ss), 1e-12f);   // matches F.normalize eps clamp

  _Float16* o = outh + (((size_t)b * H_ + h) * wpc + (w + coff)) * C_;
#pragma unroll
  for (int c8 = 0; c8 < C_ / 8; ++c8) {
    v8h tmp;
#pragma unroll
    for (int i = 0; i < 8; ++i)
      tmp[i] = (_Float16)(base[(size_t)(c8 * 8 + i) * HW] * inv);
    *(v8h*)(o + c8 * 8) = tmp;
  }
}

// ---------------------------------------------------------------------------
// Stage 2: banded-Gram correlation via v_wmma_f32_16x16x32_f16.
// Block = 256 threads = 8 waves; one block per (b,h); wave owns two 16-pixel
// w-tiles (ti outer, dy inner -> only one A set live, ~130 VGPRs). Each
// (ti,dy) computes two 16x16 Gram tiles (batched B loads + 8 chained WMMAs)
// and dumps the 9-wide band into bandLds[dy] with unconditional ds_store
// (cndmask'ed dump slot for off-band entries). 9 bands = ~84KB of the 320KB
// WGP LDS; one barrier total, then all 81 planes stream out fully coalesced.
// ---------------------------------------------------------------------------
__global__ __launch_bounds__(256) void cv_corr_kernel(const _Float16* __restrict__ f1,
                                                      const _Float16* __restrict__ f2,
                                                      float* __restrict__ out) {
  __shared__ float bandLds[ND][BANDW];   // ~84KB: per-dy 9x256 band + dump

  const int bh   = blockIdx.x;          // b*H + h
  const int b    = bh / H_;
  const int h    = bh - b * H_;
  const int wave = threadIdx.x >> 5;
  const int lane = threadIdx.x & 31;
  const int n16  = lane & 15;           // column / row-in-tile index
  const int hl   = lane >> 4;           // lane half

  const _Float16* f1row = f1 + (size_t)bh * (W_ * C_);

#pragma unroll 1
  for (int ti = 0; ti < 2; ++ti) {
    const int w0 = (wave * 2 + ti) * 16;

    // ---- A fragments (16x32 A operand; contiguous k, same order as B), dy-invariant for this tile.
    // lane: M = n16; K: [16*hl .. +15] contiguous.
    v16h a[4];
    {
      const _Float16* pa = f1row + (size_t)(w0 + n16) * C_ + 16 * hl;
#pragma unroll
      for (int kk = 0; kk < 4; ++kk) {
        v8h lo = *(const v8h*)(pa + kk * 32);
        v8h hi = *(const v8h*)(pa + kk * 32 + 8);
#pragma unroll
        for (int i = 0; i < 8; ++i) { a[kk][i] = lo[i]; a[kk][8 + i] = hi[i]; }
      }
    }

#pragma unroll 1
    for (int dy = 0; dy < ND; ++dy) {
      const int h2 = h + dy - 4;                 // uniform across the block
      v8f acc0 = {};
      v8f acc1 = {};
      if (h2 >= 0 && h2 < H_) {                  // uniform branch: EXEC stays full
        // lane's B column for column-tile t=0: padded col = (w0-4+n16) + 16
        const _Float16* pb = f2 +
            (((size_t)b * H_ + h2) * WP_ + (size_t)(w0 + 12 + n16)) * C_ + 16 * hl;

        if (ti == 0 && dy + 1 < ND && h2 + 1 < H_)   // warm next dy row in GL2
          __builtin_prefetch(pb + (size_t)(WP_ * C_), 0, 1);

        // ---- batch-load all 16 b128 fragments (clause), then 8 WMMAs
        v16h bb[8];
#pragma unroll
        for (int kk = 0; kk < 4; ++kk) {
          bb[kk]     = *(const v16h*)(pb + kk * 32);            // t=0, K-chunk kk
          bb[4 + kk] = *(const v16h*)(pb + 16 * C_ + kk * 32);  // t=1, K-chunk kk
        }
#pragma unroll
        for (int kk = 0; kk < 4; ++kk) {
          acc0 = __builtin_amdgcn_wmma_f32_16x16x32_f16(false, a[kk], false, bb[kk],
                                                        (short)0, acc0, false, false);
          acc1 = __builtin_amdgcn_wmma_f32_16x16x32_f16(false, a[kk], false, bb[4 + kk],
                                                        (short)0, acc1, false, false);
        }
      }
      // ---- dump band entries to LDS, unconditional ds_store w/ cndmask address.
      // C/D layout: lane = n + 16*(m>=8), VGPR r = m%8 holds D[m][n]; j = n+16*t.
      float* band = bandLds[dy];
#pragma unroll
      for (int r = 0; r < 8; ++r) {
        const int m = r + 8 * hl;                // output pixel w0+m
#pragma unroll
        for (int t = 0; t < 2; ++t) {
          const int dx = n16 + 16 * t - m;       // j - m
          const bool valid = (dx >= 0) && (dx <= 8);
          const int idx = valid ? dx * W_ + (w0 + m) : ND * W_ + lane;
          band[idx] = t ? acc1[r] : acc0[r];
        }
      }
    }
  }
  __syncthreads();

  // ---- coalesced output: 81 planes of 256 contiguous floats each (lane = w).
  {
    const int tt = threadIdx.x;                  // w coordinate
    const float invC = 1.0f / (float)C_;
    const size_t obase = (((size_t)b * NDISP) * H_ + h) * W_ + tt;
#pragma unroll 1
    for (int dy = 0; dy < ND; ++dy) {
#pragma unroll
      for (int d = 0; d < ND; ++d) {
        float v = bandLds[dy][d * W_ + tt] * invC;
        v = (v < 0.f) ? v * 0.1f : v;
        out[obase + (size_t)(dy * ND + d) * (H_ * W_)] = v;
      }
    }
  }
}

// ---------------------------------------------------------------------------
extern "C" void kernel_launch(void* const* d_in, const int* in_sizes, int n_in,
                              void* d_out, int out_size, void* d_ws, size_t ws_size,
                              hipStream_t stream) {
  const float* feat1 = (const float*)d_in[0];
  const float* feat2 = (const float*)d_in[1];
  float* out = (float*)d_out;

  // workspace: f1 NHWC f16 (32 MiB) + f2 padded-NHWC f16 (36 MiB)
  const size_t elems1 = (size_t)B_ * H_ * W_ * C_;
  _Float16* ws1 = (_Float16*)d_ws;
  _Float16* ws2 = ws1 + elems1;

  dim3 blk(256);
  dim3 gr(B_ * H_);                                  // 512 rows
  cv_pad_kernel<<<gr, blk, 0, stream>>>(ws2);

  dim3 g1((unsigned)((size_t)B_ * H_ * W_ / 256));   // 512 blocks, exact
  cv_norm_kernel<<<g1, blk, 0, stream>>>(feat1, ws1, W_, 0);
  cv_norm_kernel<<<g1, blk, 0, stream>>>(feat2, ws2, WP_, 16);

  cv_corr_kernel<<<gr, blk, 0, stream>>>(ws1, ws2, out);
}
